// MultiHeadAttention_55181739819367
// MI455X (gfx1250) — matrix operand order repaired and checked by reading, not compiled, not run on hardware
//
#include <hip/hip_runtime.h>

// ---------------- problem constants ----------------
#define BB      2
#define SS      2048
#define DD      1024
#define HH      16
#define DK      64
#define NWIN    256
#define MROWS   (BB * SS)              // 4096
#define MN      ((size_t)MROWS * DD)   // 4,194,304 elements
#define SP      (SS + 16)              // vT row pitch (slack for padded j-tile)
#define PITCH   560                    // LDS score row pitch (floats)
#define LN_EPS  1e-5f

typedef __bf16 bhalf;
typedef bhalf  bhalf8   __attribute__((ext_vector_type(8)));
typedef bhalf  bhalf16  __attribute__((ext_vector_type(16)));
typedef float  floatx8  __attribute__((ext_vector_type(8)));
typedef float  floatx4  __attribute__((ext_vector_type(4)));

union BF16x16 { bhalf8 h[2]; bhalf16 v; };

__device__ __forceinline__ floatx8 wmma_bf16(bhalf16 a, bhalf16 b, floatx8 c) {
    return __builtin_amdgcn_wmma_f32_16x16x32_bf16(false, a, false, b, (short)0, c, false, false);
}

// ---------------- fp32 -> bf16 convert ----------------
__global__ __launch_bounds__(256) void cvt_bf16_kernel(const float* __restrict__ src,
                                                       bhalf* __restrict__ dst, size_t n) {
    size_t i = (size_t)blockIdx.x * blockDim.x + threadIdx.x;
    size_t stride = (size_t)gridDim.x * blockDim.x;
    for (; i < n; i += stride) dst[i] = (bhalf)src[i];
}

// ---------------- W[K,N] fp32 -> W^T[N,K] bf16 ----------------
__global__ __launch_bounds__(256) void transpose_kernel(const float* __restrict__ W,
                                                        bhalf* __restrict__ WT) {
    size_t idx = (size_t)blockIdx.x * blockDim.x + threadIdx.x; // 1024*1024 threads
    int k = (int)(idx >> 10);
    int n = (int)(idx & 1023);
    WT[(size_t)n * 1024 + k] = (bhalf)W[idx];
}

// ---------------- QKV projection GEMM: one wave = 32x32 tile (2x2 WMMA block) ----
// X[4096,1024]bf16 @ WT[1024,1024]bf16 + bias, written head-major.
// mode 0: out[b][h][s][d]   (q, k)      mode 1: out[b][h][d][s] pitch SP (v^T)
__global__ __launch_bounds__(256) void gemm_qkv_kernel(const bhalf* __restrict__ X,
                                                       const bhalf* __restrict__ WT,
                                                       const float* __restrict__ bias,
                                                       bhalf* __restrict__ out, int mode) {
    const int wave = threadIdx.x >> 5;
    const int lane = threadIdx.x & 31;
    const int tile = blockIdx.x * 8 + wave;          // 4096 tiles exactly (128 x 32)
    const int itp = tile >> 5, ntp = tile & 31;
    const int m0 = itp * 32, n0 = ntp * 32;
    const int half = lane >> 4, l15 = lane & 15;

    const bhalf* arow0 = X  + (size_t)(m0 + l15) * 1024;
    const bhalf* arow1 = X  + (size_t)(m0 + 16 + l15) * 1024;
    const bhalf* brow0 = WT + (size_t)(n0 + l15) * 1024 + 16 * half;
    const bhalf* brow1 = WT + (size_t)(n0 + 16 + l15) * 1024 + 16 * half;

    floatx8 acc[2][2];
#pragma unroll
    for (int mi = 0; mi < 2; ++mi)
#pragma unroll
        for (int ni = 0; ni < 2; ++ni)
            acc[mi][ni] = (floatx8){0.f,0.f,0.f,0.f,0.f,0.f,0.f,0.f};

    for (int k = 0; k < 1024; k += 32) {
        BF16x16 a0, a1, b0, b1;
        a0.h[0] = *(const bhalf8*)(arow0 + k + 16 * half);
        a0.h[1] = *(const bhalf8*)(arow0 + k + 16 * half + 8);
        a1.h[0] = *(const bhalf8*)(arow1 + k + 16 * half);
        a1.h[1] = *(const bhalf8*)(arow1 + k + 16 * half + 8);
        b0.h[0] = *(const bhalf8*)(brow0 + k);
        b0.h[1] = *(const bhalf8*)(brow0 + k + 8);
        b1.h[0] = *(const bhalf8*)(brow1 + k);
        b1.h[1] = *(const bhalf8*)(brow1 + k + 8);
        acc[0][0] = wmma_bf16(a0.v, b0.v, acc[0][0]);
        acc[0][1] = wmma_bf16(a0.v, b1.v, acc[0][1]);
        acc[1][0] = wmma_bf16(a1.v, b0.v, acc[1][0]);
        acc[1][1] = wmma_bf16(a1.v, b1.v, acc[1][1]);
    }

#pragma unroll
    for (int mi = 0; mi < 2; ++mi) {
#pragma unroll
        for (int ni = 0; ni < 2; ++ni) {
            const int n = n0 + ni * 16 + l15;
            const float bv = bias[n];
            const int h = n >> 6, d = n & 63;
#pragma unroll
            for (int r = 0; r < 8; ++r) {
                const int m = m0 + mi * 16 + half * 8 + r;
                const int b = m >> 11;        // / S
                const int s = m & 2047;       // % S
                const float v = acc[mi][ni][r] + bv;
                size_t o;
                if (mode == 0) o = (((size_t)b * HH + h) * SS + s) * 64 + d;
                else           o = (((size_t)b * HH + h) * 64 + d) * SP + s;
                out[o] = (bhalf)v;
            }
        }
    }
}

// ---------------- banded attention: one wave per (b,h,i-tile) ----------------
__global__ __launch_bounds__(32) void attn_kernel(const bhalf* __restrict__ qh,
                                                  const bhalf* __restrict__ kh,
                                                  const bhalf* __restrict__ vT,
                                                  float* __restrict__ attn_out,
                                                  bhalf* __restrict__ ctx_out) {
    __shared__ float sm[16 * PITCH];

    const int wid = blockIdx.x;          // 0..4095
    const int it = wid & 127;
    const int h  = (wid >> 7) & 15;
    const int b  = wid >> 11;
    const int lane = threadIdx.x;
    const int l15 = lane & 15, half = lane >> 4;
    const int i0 = it * 16;

    const int jt_lo = (it - 16 > 0) ? it - 16 : 0;
    const int jt_hi = (it + 16 < 127) ? it + 16 : 127;
    const int ntiles = jt_hi - jt_lo + 1;
    const int ncols  = ntiles * 16;
    const size_t bh  = (size_t)b * HH + h;

    // ---- pass 1: scores (2 WMMAs per j-tile over DK=64) ----
    const bhalf* arow = qh + (bh * SS + i0 + l15) * 64;
    for (int t = 0; t < ntiles; ++t) {
        const int jt = jt_lo + t;
        const bhalf* brow = kh + (bh * SS + jt * 16 + l15) * 64 + 16 * half;
        floatx8 acc = {0.f,0.f,0.f,0.f,0.f,0.f,0.f,0.f};
#pragma unroll
        for (int kk = 0; kk < 64; kk += 32) {
            BF16x16 a, bf;
            a.h[0]  = *(const bhalf8*)(arow + kk + 16 * half);
            a.h[1]  = *(const bhalf8*)(arow + kk + 16 * half + 8);
            bf.h[0] = *(const bhalf8*)(brow + kk);
            bf.h[1] = *(const bhalf8*)(brow + kk + 8);
            acc = wmma_bf16(a.v, bf.v, acc);
        }
        const int j = jt * 16 + l15;
#pragma unroll
        for (int r = 0; r < 8; ++r) {
            const int i = i0 + half * 8 + r;
            int dij = i - j; if (dij < 0) dij = -dij;
            const float sval = (dij < NWIN) ? acc[r] * 0.125f : -1e30f;
            sm[(half * 8 + r) * PITCH + t * 16 + l15] = sval;
        }
    }
    __syncthreads();

    // ---- softmax per row (lanes r and r+16 split the columns) ----
    {
        const int chalf = ncols >> 1;
        const int cbeg = half * chalf;
        float* row = &sm[l15 * PITCH];
        float mx = -1e30f;
        for (int c = 0; c < chalf; ++c) mx = fmaxf(mx, row[cbeg + c]);
        mx = fmaxf(mx, __shfl_xor(mx, 16, 32));
        float ssum = 0.f;
        for (int c = 0; c < chalf; ++c) ssum += __expf(row[cbeg + c] - mx);
        ssum += __shfl_xor(ssum, 16, 32);
        const float inv = 1.0f / ssum;
        for (int c = 0; c < chalf; ++c) {
            const int cc = cbeg + c;
            row[cc] = __expf(row[cc] - mx) * inv;
        }
    }
    // zero the padded j-tile (so A-fragments of the odd tail multiply by 0)
    const int npad = (ntiles + 1) & ~1;
    if (npad * 16 > ncols) {
        for (int idx = lane; idx < 256; idx += 32)
            sm[(idx >> 4) * PITCH + ncols + (idx & 15)] = 0.f;
    }
    __syncthreads();

    // ---- stream full attn rows (band probs + zeros) with NT 128-bit stores ----
    // attn is 537 MB (>> 192 MB L2) and never re-read: bypass cache.
    const int cl0 = jt_lo * 16, cl1 = jt_lo * 16 + ncols;
    for (int m = 0; m < 16; ++m) {
        float* dst = attn_out + (bh * SS + (i0 + m)) * (size_t)SS;
        const float* src = &sm[m * PITCH];
        for (int c4 = lane; c4 < (SS >> 2); c4 += 32) {
            const int c = c4 << 2;
            floatx4 v;
#pragma unroll
            for (int e = 0; e < 4; ++e) {
                const int cc = c + e;
                v[e] = (cc >= cl0 && cc < cl1) ? src[cc - cl0] : 0.f;
            }
            __builtin_nontemporal_store(v, (floatx4*)(dst + c));
        }
    }

    // ---- ctx = P @ V : 4 d-tiles, K over padded band (32 per WMMA) ----
    const int npairs = npad >> 1;
    for (int dt = 0; dt < 4; ++dt) {
        floatx8 acc = {0.f,0.f,0.f,0.f,0.f,0.f,0.f,0.f};
        const int dglob = dt * 16 + l15;
        const bhalf* vrow = vT + (bh * 64 + dglob) * SP + jt_lo * 16 + 16 * half;
        for (int p = 0; p < npairs; ++p) {
            const int cb = p * 32;
            BF16x16 a, bf;
            const float* pr0 = &sm[l15 * PITCH + cb + 16 * half];
            const float* pr1 = &sm[l15 * PITCH + cb + 16 * half + 8];
#pragma unroll
            for (int e = 0; e < 8; ++e) { a.h[0][e] = (bhalf)pr0[e]; a.h[1][e] = (bhalf)pr1[e]; }
            bf.h[0] = *(const bhalf8*)(vrow + cb);
            bf.h[1] = *(const bhalf8*)(vrow + cb + 8);
            acc = wmma_bf16(a.v, bf.v, acc);
        }
#pragma unroll
        for (int r = 0; r < 8; ++r) {
            const int i = i0 + half * 8 + r;
            ctx_out[((size_t)b * SS + i) * 1024 + h * 64 + dglob] = (bhalf)acc[r];
        }
    }
}

// ---------------- output projection + bias + residual: 32x32 per wave ----------
__global__ __launch_bounds__(256) void oproj_kernel(const bhalf* __restrict__ ctx,
                                                    const bhalf* __restrict__ WoT,
                                                    const float* __restrict__ bo,
                                                    const float* __restrict__ Qres,
                                                    float* __restrict__ xout) {
    const int wave = threadIdx.x >> 5;
    const int lane = threadIdx.x & 31;
    const int tile = blockIdx.x * 8 + wave;          // 4096 tiles (128 x 32)
    const int itp = tile >> 5, ntp = tile & 31;
    const int m0 = itp * 32, n0 = ntp * 32;
    const int half = lane >> 4, l15 = lane & 15;

    const bhalf* arow0 = ctx + (size_t)(m0 + l15) * 1024;
    const bhalf* arow1 = ctx + (size_t)(m0 + 16 + l15) * 1024;
    const bhalf* brow0 = WoT + (size_t)(n0 + l15) * 1024 + 16 * half;
    const bhalf* brow1 = WoT + (size_t)(n0 + 16 + l15) * 1024 + 16 * half;

    floatx8 acc[2][2];
#pragma unroll
    for (int mi = 0; mi < 2; ++mi)
#pragma unroll
        for (int ni = 0; ni < 2; ++ni)
            acc[mi][ni] = (floatx8){0.f,0.f,0.f,0.f,0.f,0.f,0.f,0.f};

    for (int k = 0; k < 1024; k += 32) {
        BF16x16 a0, a1, b0, b1;
        a0.h[0] = *(const bhalf8*)(arow0 + k + 16 * half);
        a0.h[1] = *(const bhalf8*)(arow0 + k + 16 * half + 8);
        a1.h[0] = *(const bhalf8*)(arow1 + k + 16 * half);
        a1.h[1] = *(const bhalf8*)(arow1 + k + 16 * half + 8);
        b0.h[0] = *(const bhalf8*)(brow0 + k);
        b0.h[1] = *(const bhalf8*)(brow0 + k + 8);
        b1.h[0] = *(const bhalf8*)(brow1 + k);
        b1.h[1] = *(const bhalf8*)(brow1 + k + 8);
        acc[0][0] = wmma_bf16(a0.v, b0.v, acc[0][0]);
        acc[0][1] = wmma_bf16(a0.v, b1.v, acc[0][1]);
        acc[1][0] = wmma_bf16(a1.v, b0.v, acc[1][0]);
        acc[1][1] = wmma_bf16(a1.v, b1.v, acc[1][1]);
    }

#pragma unroll
    for (int mi = 0; mi < 2; ++mi) {
#pragma unroll
        for (int ni = 0; ni < 2; ++ni) {
            const int n = n0 + ni * 16 + l15;
            const float bv = bo[n];
#pragma unroll
            for (int r = 0; r < 8; ++r) {
                const size_t m = (size_t)(m0 + mi * 16 + half * 8 + r);
                xout[m * 1024 + n] = acc[mi][ni][r] + bv + Qres[m * 1024 + n];
            }
        }
    }
}

// ---------------- LayerNorm over last dim (block per row) ----------------
__global__ __launch_bounds__(256) void ln_kernel(const float* __restrict__ x,
                                                 const float* __restrict__ gamma,
                                                 const float* __restrict__ beta,
                                                 float* __restrict__ y) {
    __shared__ float red[256];
    const size_t row = blockIdx.x;
    const float* xr = x + row * 1024;
    float vals[4];
    float s = 0.f;
#pragma unroll
    for (int i = 0; i < 4; ++i) { vals[i] = xr[threadIdx.x + 256 * i]; s += vals[i]; }
    red[threadIdx.x] = s; __syncthreads();
    for (int st = 128; st > 0; st >>= 1) {
        if (threadIdx.x < st) red[threadIdx.x] += red[threadIdx.x + st];
        __syncthreads();
    }
    const float mu = red[0] * (1.0f / 1024.0f);
    __syncthreads();
    float s2 = 0.f;
#pragma unroll
    for (int i = 0; i < 4; ++i) { const float d = vals[i] - mu; s2 += d * d; }
    red[threadIdx.x] = s2; __syncthreads();
    for (int st = 128; st > 0; st >>= 1) {
        if (threadIdx.x < st) red[threadIdx.x] += red[threadIdx.x + st];
        __syncthreads();
    }
    const float rs = rsqrtf(red[0] * (1.0f / 1024.0f) + LN_EPS);
#pragma unroll
    for (int i = 0; i < 4; ++i) {
        const int c = threadIdx.x + 256 * i;
        y[row * 1024 + c] = (vals[i] - mu) * rs * gamma[c] + beta[c];
    }
}

// ---------------- host-side orchestration ----------------
extern "C" void kernel_launch(void* const* d_in, const int* in_sizes, int n_in,
                              void* d_out, int out_size, void* d_ws, size_t ws_size,
                              hipStream_t stream) {
    (void)in_sizes; (void)n_in; (void)out_size; (void)ws_size;
    const float* Qf    = (const float*)d_in[0];
    const float* Kf    = (const float*)d_in[1];
    const float* Vf    = (const float*)d_in[2];
    const float* Wq    = (const float*)d_in[3];
    const float* bq    = (const float*)d_in[4];
    const float* Wk    = (const float*)d_in[5];
    const float* bk    = (const float*)d_in[6];
    const float* Wv    = (const float*)d_in[7];
    const float* bv    = (const float*)d_in[8];
    const float* Wo    = (const float*)d_in[9];
    const float* bo    = (const float*)d_in[10];
    const float* gamma = (const float*)d_in[11];
    const float* beta  = (const float*)d_in[12];

    float* out_y    = (float*)d_out;              // [B,S,D]
    float* out_attn = out_y + MN;                 // [B,H,S,S]

    char* ws = (char*)d_ws;
    size_t off = 0;
    auto alloc = [&](size_t bytes) -> void* {
        void* p = ws + off;
        off = (off + bytes + 255) & ~(size_t)255;
        return p;
    };
    bhalf* Qb  = (bhalf*)alloc(MN * 2);
    bhalf* Kb  = (bhalf*)alloc(MN * 2);
    bhalf* Vb  = (bhalf*)alloc(MN * 2);
    bhalf* WqT = (bhalf*)alloc((size_t)1024 * 1024 * 2);
    bhalf* WkT = (bhalf*)alloc((size_t)1024 * 1024 * 2);
    bhalf* WvT = (bhalf*)alloc((size_t)1024 * 1024 * 2);
    bhalf* WoT = (bhalf*)alloc((size_t)1024 * 1024 * 2);
    bhalf* qh  = (bhalf*)alloc(MN * 2);                              // [B,H,S,64]
    bhalf* khd = (bhalf*)alloc(MN * 2);                              // [B,H,S,64]
    size_t vT_bytes = (size_t)BB * HH * 64 * SP * 2;
    bhalf* vT  = (bhalf*)alloc(vT_bytes);                            // [B,H,64,SP]
    bhalf* ctx = (bhalf*)alloc(MN * 2);                              // [B*S, H*DV]
    float* xbf = (float*)alloc(MN * 4);                              // residual sum

    // bf16 conversions of activations
    cvt_bf16_kernel<<<4096, 256, 0, stream>>>(Qf, Qb, MN);
    cvt_bf16_kernel<<<4096, 256, 0, stream>>>(Kf, Kb, MN);
    cvt_bf16_kernel<<<4096, 256, 0, stream>>>(Vf, Vb, MN);
    // weight transposes to [N,K] bf16
    transpose_kernel<<<4096, 256, 0, stream>>>(Wq, WqT);
    transpose_kernel<<<4096, 256, 0, stream>>>(Wk, WkT);
    transpose_kernel<<<4096, 256, 0, stream>>>(Wv, WvT);
    transpose_kernel<<<4096, 256, 0, stream>>>(Wo, WoT);
    // zero vT (slack columns must be benign for padded band tiles)
    hipMemsetAsync(vT, 0, vT_bytes, stream);

    // QKV projections (WMMA GEMMs, 32x32 per wave)
    gemm_qkv_kernel<<<512, 256, 0, stream>>>(Qb, WqT, bq, qh,  0);
    gemm_qkv_kernel<<<512, 256, 0, stream>>>(Kb, WkT, bk, khd, 0);
    gemm_qkv_kernel<<<512, 256, 0, stream>>>(Vb, WvT, bv, vT,  1);

    // banded attention + softmax + ctx (WMMA)
    attn_kernel<<<4096, 32, 0, stream>>>(qh, khd, vT, out_attn, ctx);

    // output projection + residual (WMMA, 32x32 per wave), then LayerNorm
    oproj_kernel<<<512, 256, 0, stream>>>(ctx, WoT, bo, Qf, xbf);
    ln_kernel<<<4096, 256, 0, stream>>>(xbf, gamma, beta, out_y);
}
